// InterleavedPhaseChannelizer_21260088115665
// MI455X (gfx1250) — matrix operand order repaired and checked by reading, not compiled, not run on hardware
//
#include <hip/hip_runtime.h>
#include <cstdint>
#include <cstddef>

#define T_DIM 4096
#define B_DIM 4
#define E_DIM 1024
#define M_DIM 512
#define EPS_K 1e-6f

typedef _Float16 v8h  __attribute__((ext_vector_type(8)));
typedef _Float16 v16h __attribute__((ext_vector_type(16)));
typedef float    v8f  __attribute__((ext_vector_type(8)));

// Bump kernel value before row-normalization (0 below diagonal).
__device__ __forceinline__ float ipc_bump(int i, int j) {
    if (j < i) return 0.0f;
    float diff    = (float)(j - i);
    float horizon = fmaxf((float)T_DIM - (float)i, 1.0f);
    float u = diff / horizon;
    if (u >= 1.0f - EPS_K) return EPS_K;       // clamped branch of reference
    float den = 1.0f - u * u + EPS_K;
    return __expf(1.0f - 1.0f / den);
}

// One block per row i: reduce row sum, then write normalized f16 row.
__global__ __launch_bounds__(256) void ipc_build_k(_Float16* __restrict__ Kf16) {
    const int i   = blockIdx.x;
    const int tid = threadIdx.x;
    __shared__ float red[256];

    float s = 0.0f;
    for (int j = tid; j < T_DIM; j += 256) s += ipc_bump(i, j);
    red[tid] = s;
    __syncthreads();
    for (int off = 128; off > 0; off >>= 1) {
        if (tid < off) red[tid] += red[tid + off];
        __syncthreads();
    }
    const float inv = 1.0f / fmaxf(red[0], EPS_K);

    _Float16* row = Kf16 + (size_t)i * T_DIM;
    for (int j = tid; j < T_DIM; j += 256)
        row[j] = (_Float16)(ipc_bump(i, j) * inv);
}

// xcT[((b*M)+m)*T + t] = (f16) x[b][t][2m]   (B-operand staging, contiguous in k)
__global__ __launch_bounds__(256) void ipc_build_xct(const float* __restrict__ x,
                                                     _Float16* __restrict__ xcT) {
    size_t idx = (size_t)blockIdx.x * 256 + threadIdx.x;
    if (idx >= (size_t)B_DIM * M_DIM * T_DIM) return;
    int t = (int)(idx % T_DIM);
    int m = (int)((idx / T_DIM) % M_DIM);
    int b = (int)(idx / ((size_t)T_DIM * M_DIM));
    float v = x[((size_t)(b * T_DIM + t)) * E_DIM + 2 * m];
    xcT[idx] = (_Float16)v;
}

__global__ void ipc_gate(const float* __restrict__ graw, float* __restrict__ gate) {
    int m = blockIdx.x * blockDim.x + threadIdx.x;
    if (m < M_DIM) {
        float g = graw[m];
        gate[m] = (g > 20.0f) ? g : log1pf(__expf(g));   // softplus
    }
}

// Full-bandwidth copy x -> out (odd channels overwritten by WMMA kernel).
__global__ __launch_bounds__(256) void ipc_copy(const float4* __restrict__ x,
                                                float4* __restrict__ out, int n4) {
    int stride = gridDim.x * blockDim.x;
    for (int i = blockIdx.x * blockDim.x + threadIdx.x; i < n4; i += stride)
        out[i] = x[i];
}

// A fragment gather in contiguous k order (same order as B):
// lane<16: row=l15, K {k..k+15}; lane>=16: K {k+16..k+31}.
__device__ __forceinline__ v16h ipc_load_a(const _Float16* p, int k) {
    union { v16h v; v8h h[2]; } a;
    a.h[0] = *(const v8h*)(p + k);
    a.h[1] = *(const v8h*)(p + k + 8);
    return a.v;
}

__device__ __forceinline__ v8f ipc_mma(const v16h& a, const v16h& b, const v8f& c) {
    return __builtin_amdgcn_wmma_f32_16x16x32_f16(false, a, false, b,
                                                  (short)0, c, false, false);
}

// phi = K @ xc per batch via v_wmma_f32_16x16x32_f16; write odd channels.
// Block: 8 waves as 2(rows)x4(cols); each wave owns a 32x32 output tile
// (2 A-frags x 2 B-frags -> 4 accumulators, 4 WMMAs per k-step). Loads are
// loop-local so `#pragma unroll 2` lets the scheduler pipeline iteration
// k+32's load clause above iteration k's WMMAs with no register copies.
__global__ __launch_bounds__(256) void ipc_wmma(const _Float16* __restrict__ Kf16,
                                                const _Float16* __restrict__ xcT,
                                                const float* __restrict__ gate,
                                                float* __restrict__ out) {
    const int lane = threadIdx.x & 31;
    const int wave = threadIdx.x >> 5;
    const int wr   = wave & 1;                      // wave row group (0..1)
    const int wc   = wave >> 1;                     // wave col group (0..3)
    const int r0   = blockIdx.y * 64 + wr * 32;     // wave's first output row
    const int c0   = blockIdx.x * 128 + wc * 32;    // wave's first output col
    const int b    = blockIdx.z;
    const int half = lane >> 4;                     // 0: lanes 0-15, 1: lanes 16-31
    const int l15  = lane & 15;

    const _Float16* arow0 = Kf16 + (size_t)(r0 + l15) * T_DIM + half * 16;
    const _Float16* arow1 = arow0 + (size_t)16 * T_DIM;
    // B fragment (32x16 f16): column c + l15, 16 contiguous K values at k + half*16.
    const _Float16* bcol0 = xcT + ((size_t)(b * M_DIM + c0 + l15)) * T_DIM + half * 16;
    const _Float16* bcol1 = bcol0 + (size_t)16 * T_DIM;

    v8f acc00 = {}, acc01 = {}, acc10 = {}, acc11 = {};

    // Rows covered: [r0, r0+32). Tiles with k+32 <= r0 are all-zero (triangular K).
    const int kstart = r0;                          // r0 is a multiple of 32

#pragma unroll 2
    for (int k = kstart; k < T_DIM; k += 32) {
        v16h a0 = ipc_load_a(arow0, k);
        v16h a1 = ipc_load_a(arow1, k);
        v16h b0 = *(const v16h*)(bcol0 + k);
        v16h b1 = *(const v16h*)(bcol1 + k);
        acc00 = ipc_mma(a0, b0, acc00);
        acc01 = ipc_mma(a0, b1, acc01);
        acc10 = ipc_mma(a1, b0, acc10);
        acc11 = ipc_mma(a1, b1, acc11);
    }

    // C/D layout: VGPR v -> row rbase + v + (half ? 8 : 0), col c + l15.
    const float g0 = gate[c0 + l15];
    const float g1 = gate[c0 + 16 + l15];
    const int   m0 = c0 + l15;
    const int   m1 = c0 + 16 + l15;

    const v8f* accs[4]  = { &acc00, &acc01, &acc10, &acc11 };
    const int  rows[4]  = { r0, r0, r0 + 16, r0 + 16 };
    const int  cols[4]  = { m0, m1, m0, m1 };
    const float gs[4]   = { g0, g1, g0, g1 };
#pragma unroll
    for (int t = 0; t < 4; ++t) {
        const int rbase = rows[t] + half * 8;
        const v8f& a = *accs[t];
#pragma unroll
        for (int v = 0; v < 8; ++v) {
            out[((size_t)(b * T_DIM + rbase + v)) * E_DIM + 2 * cols[t] + 1] =
                a[v] * gs[t];
        }
    }
}

extern "C" void kernel_launch(void* const* d_in, const int* in_sizes, int n_in,
                              void* d_out, int out_size, void* d_ws, size_t ws_size,
                              hipStream_t stream) {
    (void)in_sizes; (void)n_in; (void)out_size; (void)ws_size;
    const float* x        = (const float*)d_in[0];
    // d_in[1]: triangular bool mask — structurally known, not read.
    const float* gate_raw = (const float*)d_in[2];
    float*       out      = (float*)d_out;

    char* ws = (char*)d_ws;
    const size_t K_BYTES  = (size_t)T_DIM * T_DIM * sizeof(_Float16);         // 33.5 MB
    const size_t XT_BYTES = (size_t)B_DIM * M_DIM * T_DIM * sizeof(_Float16); // 16.8 MB
    _Float16* Kf16 = (_Float16*)ws;
    _Float16* xcT  = (_Float16*)(ws + K_BYTES);
    float*    gate = (float*)(ws + K_BYTES + XT_BYTES);

    ipc_build_k<<<T_DIM, 256, 0, stream>>>(Kf16);

    {
        size_t n = (size_t)B_DIM * M_DIM * T_DIM;
        ipc_build_xct<<<(unsigned)((n + 255) / 256), 256, 0, stream>>>(x, xcT);
    }

    ipc_gate<<<(M_DIM + 255) / 256, 256, 0, stream>>>(gate_raw, gate);

    {
        int n4 = B_DIM * T_DIM * E_DIM / 4;
        ipc_copy<<<2048, 256, 0, stream>>>((const float4*)x, (float4*)out, n4);
    }

    dim3 grid(M_DIM / 128, T_DIM / 64, B_DIM);
    ipc_wmma<<<grid, 256, 0, stream>>>(Kf16, xcT, gate, out);
}
